// EncoderBlock_32933809226478
// MI455X (gfx1250) — matrix operand order repaired and checked by reading, not compiled, not run on hardware
//
#include <hip/hip_runtime.h>
#include <hip/hip_bf16.h>
#include <math.h>

typedef _Float16 h16;
typedef __attribute__((ext_vector_type(16))) _Float16 v16h;
typedef __attribute__((ext_vector_type(8)))  float    v8f;
typedef __attribute__((ext_vector_type(4)))  unsigned u32x4;
typedef int v4i_vec __attribute__((__vector_size__(4 * sizeof(int))));

// trivially-copyable union for building WMMA fragments
union F16Frag { v16h v; unsigned u[8]; u32x4 q[2]; };

#define NB   16
#define NS   384
#define ND   512
#define NH   8
#define NHD  64
#define NL   4
#define NK   7
#define NTOK (NB * NS)   // 6144

// CDNA5 async global->LDS DMA path (ASYNCcnt), with sync fallback
#if defined(__gfx1250__) && \
    __has_builtin(__builtin_amdgcn_global_load_async_to_lds_b128) && \
    __has_builtin(__builtin_amdgcn_s_wait_asynccnt)
#define USE_ASYNC_LDS 1
#else
#define USE_ASYNC_LDS 0
#endif

__device__ __forceinline__ void cp_lds128(h16* ldst, const h16* gsrc) {
#if USE_ASYNC_LDS
    __builtin_amdgcn_global_load_async_to_lds_b128(
        (__attribute__((address_space(1))) v4i_vec*)gsrc,
        (__attribute__((address_space(3))) v4i_vec*)ldst,
        0, 0);
#else
    *(u32x4*)ldst = *(const u32x4*)gsrc;
#endif
}

// ---------------------------------------------------------------------------
// f32 -> f16 conversion (weights)
// ---------------------------------------------------------------------------
__global__ void cvt_f32_f16_kernel(const float* __restrict__ src,
                                   h16* __restrict__ dst, int n) {
    int i = blockIdx.x * 256 + threadIdx.x;
    if (i < n) dst[i] = (h16)src[i];
}

// ---------------------------------------------------------------------------
// out = x + pos_enc ; res = out ; cur = LayerNorm(out)
// one block per token, 256 threads, 2 elements per thread
// ---------------------------------------------------------------------------
__global__ __launch_bounds__(256) void addpos_ln_kernel(
    const float* __restrict__ x, const float* __restrict__ g,
    const float* __restrict__ bb, float* __restrict__ res,
    float* __restrict__ cur) {
    const int t = blockIdx.x;
    const int s = t % NS;
    const int tid = threadIdx.x;

    float v[2];
#pragma unroll
    for (int e = 0; e < 2; ++e) {
        const int d = tid + e * 256;
        const float fr  = powf(10000.f, (2.f * (float)d) / (float)ND);
        const float ang = (float)s / fr;
        const float pe  = (d & 1) ? cosf(ang) : sinf(ang);
        v[e] = x[(size_t)t * ND + d] + pe;
        res[(size_t)t * ND + d] = v[e];
    }
    float s1 = v[0] + v[1];
    float s2 = v[0] * v[0] + v[1] * v[1];
#pragma unroll
    for (int o = 16; o; o >>= 1) {
        s1 += __shfl_xor(s1, o, 32);
        s2 += __shfl_xor(s2, o, 32);
    }
    __shared__ float sa[8], sb[8];
    if ((tid & 31) == 0) { sa[tid >> 5] = s1; sb[tid >> 5] = s2; }
    __syncthreads();
    float t1 = 0.f, t2 = 0.f;
#pragma unroll
    for (int i = 0; i < 8; ++i) { t1 += sa[i]; t2 += sb[i]; }
    const float mean = t1 * (1.f / (float)ND);
    const float var  = t2 * (1.f / (float)ND) - mean * mean;
    const float rs   = rsqrtf(var + 1e-5f);
#pragma unroll
    for (int e = 0; e < 2; ++e) {
        const int d = tid + e * 256;
        cur[(size_t)t * ND + d] = (v[e] - mean) * rs * g[d] + bb[d];
    }
}

// ---------------------------------------------------------------------------
// generic LayerNorm: outF = LN(in)*g+b ; outH = (h16)outF
// ---------------------------------------------------------------------------
__global__ __launch_bounds__(256) void ln_kernel(
    const float* __restrict__ in, const float* __restrict__ g,
    const float* __restrict__ bb, float* __restrict__ outF,
    h16* __restrict__ outH) {
    const int t = blockIdx.x;
    const int tid = threadIdx.x;
    float v[2];
#pragma unroll
    for (int e = 0; e < 2; ++e) v[e] = in[(size_t)t * ND + tid + e * 256];

    float s1 = v[0] + v[1];
    float s2 = v[0] * v[0] + v[1] * v[1];
#pragma unroll
    for (int o = 16; o; o >>= 1) {
        s1 += __shfl_xor(s1, o, 32);
        s2 += __shfl_xor(s2, o, 32);
    }
    __shared__ float sa[8], sb[8];
    if ((tid & 31) == 0) { sa[tid >> 5] = s1; sb[tid >> 5] = s2; }
    __syncthreads();
    float t1 = 0.f, t2 = 0.f;
#pragma unroll
    for (int i = 0; i < 8; ++i) { t1 += sa[i]; t2 += sb[i]; }
    const float mean = t1 * (1.f / (float)ND);
    const float var  = t2 * (1.f / (float)ND) - mean * mean;
    const float rs   = rsqrtf(var + 1e-5f);
#pragma unroll
    for (int e = 0; e < 2; ++e) {
        const int d = tid + e * 256;
        const float o = (v[e] - mean) * rs * g[d] + bb[d];
        outF[(size_t)t * ND + d] = o;
        outH[(size_t)t * ND + d] = (h16)o;
    }
}

// ---------------------------------------------------------------------------
// depthwise conv1d over sequence, K=7, pad 3, f32 in -> f16 out
// ---------------------------------------------------------------------------
__global__ void dwconv_kernel(const float* __restrict__ cur,
                              const float* __restrict__ dw,  // [D*K] this layer
                              h16* __restrict__ convh) {
    const int idx = blockIdx.x * 256 + threadIdx.x;  // token*512 + d
    const int d = idx & (ND - 1);
    const int t = idx >> 9;
    const int s = t % NS;
    const float* wr = dw + d * NK;
    const float* base = cur + (size_t)(t - s) * ND + d;  // batch start
    float acc = 0.f;
#pragma unroll
    for (int k = 0; k < NK; ++k) {
        const int ss = s + k - 3;
        if (ss >= 0 && ss < NS) acc += base[(size_t)ss * ND] * wr[k];
    }
    convh[idx] = (h16)acc;
}

// ---------------------------------------------------------------------------
// WMMA GEMM: Y[m,n] = sum_k X[m,k]*W[n,k] + bias[n]  (M=6144,N=512,K=512)
// block tile 64x64, 4 waves (2x2), each wave 32x32 (2x2 WMMA 16x16x32 tiles)
// double-buffered LDS; staging via async global->LDS DMA (ASYNCcnt) when
// available, so tile k+1 streams in underneath the WMMAs of tile k.
// mode 0: outF = relu(y) + res     (conv pointwise, FF)
// mode 1: outH = f16(y*scale) into [B,H,S,HD] (vtrans=0) or [B,H,HD,S] (=1)
// mode 2: outF = y + res           (attention output projection)
// ---------------------------------------------------------------------------
__global__ __launch_bounds__(128) void gemm_kernel(
    const h16* __restrict__ X, const h16* __restrict__ W,
    const float* __restrict__ bias, const float* __restrict__ resv,
    float* __restrict__ outF, h16* __restrict__ outH,
    float scale, int mode, int vtrans) {
    __shared__ h16 Xs[2][64][40];   // 8-half pad: 80B row stride, conflict-free
    __shared__ h16 Ws[2][64][40];

    const int tid  = threadIdx.x;
    const int lane = tid & 31;
    const int w    = tid >> 5;
    const int wm   = w >> 1, wn = w & 1;
    const int l15  = lane & 15, lhi = lane >> 4;
    const int m0   = blockIdx.x * 64, n0 = blockIdx.y * 64;

    const int row = tid >> 1;          // 0..63
    const int seg = (tid & 1) * 16;    // 0 | 16
    const h16* gxb = X + (size_t)(m0 + row) * ND + seg;
    const h16* gwb = W + (size_t)(n0 + row) * ND + seg;

    v8f acc[2][2];
    acc[0][0] = {}; acc[0][1] = {}; acc[1][0] = {}; acc[1][1] = {};

    // stage K-tile kt into LDS buffer bufi (4 x b128 per thread)
    auto stage = [&](int kt, int bufi) {
        const h16* gx = gxb + kt * 32;
        const h16* gw = gwb + kt * 32;
        cp_lds128(&Xs[bufi][row][seg],     gx);
        cp_lds128(&Xs[bufi][row][seg + 8], gx + 8);
        cp_lds128(&Ws[bufi][row][seg],     gw);
        cp_lds128(&Ws[bufi][row][seg + 8], gw + 8);
    };

    stage(0, 0);
    for (int kt = 0; kt < 16; ++kt) {
        const int cb = kt & 1;
        if (kt < 15) {
            stage(kt + 1, cb ^ 1);
#if !USE_ASYNC_LDS
            __builtin_prefetch(gxb + (kt + 2) * 32, 0, 1);  // harmless at edge
            __builtin_prefetch(gwb + (kt + 2) * 32, 0, 1);
#endif
        }
#if USE_ASYNC_LDS
        // retire tile kt's 4 DMAs (in order); tile kt+1's 4 stay in flight
        if (kt < 15) __builtin_amdgcn_s_wait_asynccnt(4);
        else         __builtin_amdgcn_s_wait_asynccnt(0);
#endif
        __syncthreads();   // tile kt visible to all waves

        F16Frag a[2], bf[2];
#pragma unroll
        for (int mt = 0; mt < 2; ++mt) {
            const int m = wm * 32 + mt * 16 + l15;
            const int hi8 = lhi * 8;
#pragma unroll
            for (int pp = 0; pp < 8; ++pp) {
                const int kk = (pp < 4 ? 2 * pp : 2 * pp + 8) + hi8;
                a[mt].u[pp] = *(const unsigned*)&Xs[cb][m][kk];
            }
        }
#pragma unroll
        for (int nt = 0; nt < 2; ++nt) {
            const int n = wn * 32 + nt * 16 + l15;
            const int ko = lhi * 8;
            bf[nt].q[0] = *(const u32x4*)&Ws[cb][n][ko];
            bf[nt].q[1] = *(const u32x4*)&Ws[cb][n][ko + 16];
        }
#pragma unroll
        for (int mt = 0; mt < 2; ++mt)
#pragma unroll
            for (int nt = 0; nt < 2; ++nt)
                acc[mt][nt] = __builtin_amdgcn_wmma_f32_16x16x32_f16(
                    false, a[mt].v, false, bf[nt].v, (short)0, acc[mt][nt],
                    false, false);

        __syncthreads();   // all reads of buffer cb done before it is refilled
    }

    // epilogue
#pragma unroll
    for (int mt = 0; mt < 2; ++mt)
#pragma unroll
        for (int nt = 0; nt < 2; ++nt) {
            const int n  = n0 + wn * 32 + nt * 16 + l15;
            const int mb = m0 + wm * 32 + mt * 16 + lhi * 8;
            const float bn = bias[n];
#pragma unroll
            for (int r = 0; r < 8; ++r) {
                const int m = mb + r;
                float y = acc[mt][nt][r] + bn;
                if (mode == 0) {
                    y = y > 0.f ? y : 0.f;
                    outF[(size_t)m * ND + n] = y + resv[(size_t)m * ND + n];
                } else if (mode == 2) {
                    outF[(size_t)m * ND + n] = y + resv[(size_t)m * ND + n];
                } else {  // qkv head layouts
                    const int b = m / NS, s = m % NS;
                    const int h = n >> 6, d = n & 63;
                    const float yv = y * scale;
                    if (vtrans)
                        outH[((size_t)(b * NH + h) * NHD + d) * NS + s] = (h16)yv;
                    else
                        outH[((size_t)(b * NH + h) * NS + s) * NHD + d] = (h16)yv;
                }
            }
        }
}

// ---------------------------------------------------------------------------
// attention: per block one (b,h, 32 q-rows) tile; 2 waves, 16 q-rows each.
// scores via WMMA into 48KB LDS, masked softmax (wave32 shfl reductions),
// in-place f32->f16 alias (per-row region), then P@V via WMMA.
// q is pre-scaled by 1/sqrt(HD); v stored transposed [B,H,HD,S].
// ---------------------------------------------------------------------------
__global__ __launch_bounds__(64) void attn_kernel(
    const h16* __restrict__ qh, const h16* __restrict__ kh,
    const h16* __restrict__ vt, const int* __restrict__ mask,
    h16* __restrict__ atth) {
    __shared__ float eL[32 * NS];          // 48KB scores
    h16* pL = (h16*)eL;                    // f16 alias, row stride 768 halfs

    const int tid  = threadIdx.x;
    const int lane = tid & 31;
    const int w    = tid >> 5;             // 0..1
    const int l15  = lane & 15, lhi = lane >> 4;
    const int qt   = blockIdx.x;           // 0..11
    const int bh   = blockIdx.y;           // 0..127
    const int b    = bh >> 3, h = bh & 7;
    const int qbase = qt * 32;
    const int rowb  = w * 16;

    // Q fragments for this wave's 16 rows (K=64 -> two 16x32 frags)
    F16Frag aq[2];
    {
        const int m = qbase + rowb + l15;
        const size_t base = ((size_t)bh * NS + m) * NHD;
        const int hi8 = lhi * 8;
#pragma unroll
        for (int kk = 0; kk < 2; ++kk)
#pragma unroll
            for (int pp = 0; pp < 8; ++pp) {
                const int k = kk * 32 + (pp < 4 ? 2 * pp : 2 * pp + 8) + hi8;
                aq[kk].u[pp] = *(const unsigned*)(qh + base + k);
            }
    }

    // scores (q already scaled), masked, to LDS
    for (int nt = 0; nt < 24; ++nt) {
        v8f e = {};
        const int n = nt * 16 + l15;
        const size_t kb = ((size_t)bh * NS + n) * NHD + lhi * 8;
#pragma unroll
        for (int kk = 0; kk < 2; ++kk) {
            F16Frag bk;
            bk.q[0] = *(const u32x4*)(kh + kb + kk * 32);
            bk.q[1] = *(const u32x4*)(kh + kb + kk * 32 + 16);
            e = __builtin_amdgcn_wmma_f32_16x16x32_f16(
                false, aq[kk].v, false, bk.v, (short)0, e, false, false);
        }
        const bool msk = (mask[b * NS + n] == 1);
        const int mr = rowb + lhi * 8;
#pragma unroll
        for (int r = 0; r < 8; ++r)
            eL[(mr + r) * NS + n] = msk ? -1e10f : e[r];
    }

    // softmax over this wave's rows (no cross-wave sharing -> no barrier)
    for (int rr = 0; rr < 16; ++rr) {
        float* er = eL + (size_t)(rowb + rr) * NS;
        float mx = -3.4e38f;
        for (int j = lane; j < NS; j += 32) mx = fmaxf(mx, er[j]);
#pragma unroll
        for (int o = 16; o; o >>= 1) mx = fmaxf(mx, __shfl_xor(mx, o, 32));
        float sm = 0.f;
        for (int j = lane; j < NS; j += 32) {
            const float tt = __expf(er[j] - mx);
            er[j] = tt;
            sm += tt;
        }
#pragma unroll
        for (int o = 16; o; o >>= 1) sm += __shfl_xor(sm, o, 32);
        const float inv = 1.f / sm;
        h16* pr = pL + (size_t)(rowb + rr) * 768;
        for (int j = lane; j < NS; j += 32) pr[j] = (h16)(er[j] * inv);
    }

    // out = P @ V   (K loop over 384 keys in 12 chunks of 32)
    v8f o[4];
    o[0] = {}; o[1] = {}; o[2] = {}; o[3] = {};
    for (int nc = 0; nc < 12; ++nc) {
        F16Frag ap;
        const int prow = rowb + l15;
        const int hi8 = lhi * 8;
#pragma unroll
        for (int pp = 0; pp < 8; ++pp) {
            const int k = nc * 32 + (pp < 4 ? 2 * pp : 2 * pp + 8) + hi8;
            ap.u[pp] = *(const unsigned*)(pL + (size_t)prow * 768 + k);
        }
#pragma unroll
        for (int dt = 0; dt < 4; ++dt) {
            F16Frag bv;
            const int d = dt * 16 + l15;
            const size_t vb = ((size_t)bh * NHD + d) * NS + nc * 32 + lhi * 8;
            bv.q[0] = *(const u32x4*)(vt + vb);
            bv.q[1] = *(const u32x4*)(vt + vb + 16);
            o[dt] = __builtin_amdgcn_wmma_f32_16x16x32_f16(
                false, ap.v, false, bv.v, (short)0, o[dt], false, false);
        }
    }

    // store to [BS, D] (f16) for output projection
#pragma unroll
    for (int dt = 0; dt < 4; ++dt) {
        const int d  = dt * 16 + l15;
        const int mb = qbase + rowb + lhi * 8;
#pragma unroll
        for (int r = 0; r < 8; ++r)
            atth[((size_t)(b * NS + mb + r)) * ND + h * NHD + d] = (h16)o[dt][r];
    }
}

// ---------------------------------------------------------------------------
extern "C" void kernel_launch(void* const* d_in, const int* in_sizes, int n_in,
                              void* d_out, int out_size, void* d_ws, size_t ws_size,
                              hipStream_t stream) {
    const float* x         = (const float*)d_in[0];
    const int*   mask      = (const int*)  d_in[1];
    const float* dw_w      = (const float*)d_in[2];
    const float* pw_w      = (const float*)d_in[3];
    const float* pw_b      = (const float*)d_in[4];
    const float* conv_ln_g = (const float*)d_in[5];
    const float* conv_ln_b = (const float*)d_in[6];
    const float* pos_ln_g  = (const float*)d_in[7];
    const float* pos_ln_b  = (const float*)d_in[8];
    const float* wq        = (const float*)d_in[9];
    const float* bq        = (const float*)d_in[10];
    const float* wk        = (const float*)d_in[11];
    const float* bk        = (const float*)d_in[12];
    const float* wv        = (const float*)d_in[13];
    const float* bv        = (const float*)d_in[14];
    const float* wo        = (const float*)d_in[15];
    const float* bo        = (const float*)d_in[16];
    const float* ff_ln_g   = (const float*)d_in[17];
    const float* ff_ln_b   = (const float*)d_in[18];
    const float* ff_w      = (const float*)d_in[19];
    const float* ff_b      = (const float*)d_in[20];
    float* out = (float*)d_out;

    // workspace carving
    char* p = (char*)d_ws;
    auto carve = [&](size_t bytes) {
        void* r = (void*)p;
        p += (bytes + 255) & ~(size_t)255;
        return r;
    };
    const size_t NE = (size_t)NTOK * ND;       // 3,145,728 elements
    float* res   = (float*)carve(NE * 4);
    float* cur   = (float*)carve(NE * 4);
    h16*   curh  = (h16*)  carve(NE * 2);
    h16*   convh = (h16*)  carve(NE * 2);
    h16*   qh    = (h16*)  carve(NE * 2);
    h16*   khb   = (h16*)  carve(NE * 2);
    h16*   vt    = (h16*)  carve(NE * 2);
    h16*   atth  = (h16*)  carve(NE * 2);
    h16*   pw16  = (h16*)  carve((size_t)NL * ND * ND * 2);
    h16*   wq16  = (h16*)  carve((size_t)ND * ND * 2);
    h16*   wk16  = (h16*)  carve((size_t)ND * ND * 2);
    h16*   wv16  = (h16*)  carve((size_t)ND * ND * 2);
    h16*   wo16  = (h16*)  carve((size_t)ND * ND * 2);
    h16*   ff16  = (h16*)  carve((size_t)ND * ND * 2);

    const int WN = ND * ND;  // 262144
    cvt_f32_f16_kernel<<<(NL * WN + 255) / 256, 256, 0, stream>>>(pw_w, pw16, NL * WN);
    cvt_f32_f16_kernel<<<(WN + 255) / 256, 256, 0, stream>>>(wq, wq16, WN);
    cvt_f32_f16_kernel<<<(WN + 255) / 256, 256, 0, stream>>>(wk, wk16, WN);
    cvt_f32_f16_kernel<<<(WN + 255) / 256, 256, 0, stream>>>(wv, wv16, WN);
    cvt_f32_f16_kernel<<<(WN + 255) / 256, 256, 0, stream>>>(wo, wo16, WN);
    cvt_f32_f16_kernel<<<(WN + 255) / 256, 256, 0, stream>>>(ff_w, ff16, WN);

    const dim3 gemmGrid(NTOK / 64, ND / 64);  // 96 x 8

    // pos-enc add + pos LayerNorm
    addpos_ln_kernel<<<NTOK, 256, 0, stream>>>(x, pos_ln_g, pos_ln_b, res, cur);

    // conv stack
    for (int i = 0; i < NL; ++i) {
        dwconv_kernel<<<(int)(NE / 256), 256, 0, stream>>>(
            cur, dw_w + (size_t)i * ND * NK, convh);
        gemm_kernel<<<gemmGrid, 128, 0, stream>>>(
            convh, pw16 + (size_t)i * WN, pw_b + (size_t)i * ND, res, res,
            (h16*)nullptr, 1.f, 0, 0);
        ln_kernel<<<NTOK, 256, 0, stream>>>(
            res, conv_ln_g + (size_t)i * ND, conv_ln_b + (size_t)i * ND, cur, curh);
    }

    // attention
    gemm_kernel<<<gemmGrid, 128, 0, stream>>>(curh, wq16, bq, nullptr, nullptr,
                                              qh, 0.125f, 1, 0);
    gemm_kernel<<<gemmGrid, 128, 0, stream>>>(curh, wk16, bk, nullptr, nullptr,
                                              khb, 1.f, 1, 0);
    gemm_kernel<<<gemmGrid, 128, 0, stream>>>(curh, wv16, bv, nullptr, nullptr,
                                              vt, 1.f, 1, 1);
    attn_kernel<<<dim3(NS / 32, NB * NH), 64, 0, stream>>>(qh, khb, vt, mask, atth);
    gemm_kernel<<<gemmGrid, 128, 0, stream>>>(atth, wo16, bo, res, res,
                                              (h16*)nullptr, 1.f, 2, 0);

    // feed-forward
    ln_kernel<<<NTOK, 256, 0, stream>>>(res, ff_ln_g, ff_ln_b, cur, curh);
    gemm_kernel<<<gemmGrid, 128, 0, stream>>>(curh, ff16, ff_b, res, out,
                                              (h16*)nullptr, 1.f, 0, 0);
}
